// DeepSeekMoE_41523743818317
// MI455X (gfx1250) — compile-verified, matrix operand order checked by reading, not run on hardware
//
#include <hip/hip_runtime.h>

// ---------------- problem constants ----------------
#define TTOT 4096        // B*T tokens
#define DDIM 2048        // model dim
#define HDIM 1408        // expert hidden dim
#define NEXP 16          // routed experts
#define NSH  2           // shared experts
// ---------------- GEMM tiling ----------------------
#define BM 128
#define BN 128
#define BK 32
#define LDT 40           // padded LDS row length (bf16 elems) -> 80B rows, kills bank conflicts
#define BBUF (BM * LDT)               // B tile offset inside a buffer
#define BUFSZ ((BM + BN) * LDT)       // one double-buffer slice (ushorts)
#define MAXTILES 80      // sum ceil(c_e/128) <= 8192/128 + 16
#define MAXSLOTS 10240   // 8192 assignments + padding, rounded up
#define ROUTER_BLOCKS (TTOT/8)

typedef __attribute__((ext_vector_type(16))) __bf16 v16bf;
typedef __attribute__((ext_vector_type(8)))  float  v8f;

struct alignas(16) U4 { unsigned x, y, z, w; };
union Frag { v16bf v; U4 q[2]; };

static __device__ __forceinline__ unsigned short f2bf(float f) {
  unsigned u = __float_as_uint(f);
  u += 0x7FFFu + ((u >> 16) & 1u);          // round-to-nearest-even
  return (unsigned short)(u >> 16);
}

// RNE-round two floats and merge their high halves with one v_perm_b32:
// result = bf16(hi) << 16 | bf16(lo)
static __device__ __forceinline__ unsigned pack_bf16(float lo, float hi) {
  unsigned ul = __float_as_uint(lo), uh = __float_as_uint(hi);
  ul += 0x7FFFu + ((ul >> 16) & 1u);
  uh += 0x7FFFu + ((uh >> 16) & 1u);
  return __builtin_amdgcn_perm(uh, ul, 0x07060302u);  // {uh[3],uh[2],ul[3],ul[2]}
}

// fast SiLU: x * rcp(1+exp(-x)) -> v_exp_f32 + v_rcp_f32 + v_mul (no IEEE div chain)
static __device__ __forceinline__ float silu(float x) {
  return x * __builtin_amdgcn_rcpf(1.f + __expf(-x));
}

// Async global->LDS copy of 16 bytes (tracked by ASYNCcnt, no VGPR round trip).
// dst: generic shared pointer (low 32 bits == LDS byte offset per aperture rules)
static __device__ __forceinline__ void async_cp16(void* dst, const void* src) {
  unsigned ldsoff = (unsigned)(size_t)dst;
  asm volatile("global_load_async_to_lds_b128 %0, %1, off"
               :: "v"(ldsoff), "v"(src) : "memory");
}
static __device__ __forceinline__ void wait_async0() {
  asm volatile("s_wait_asynccnt 0x0" ::: "memory");
}

// Load one 16x32 bf16 WMMA fragment from an LDS tile (row-major, LDT stride).
// Per ISA 16-bit A layout: lanes 0-15 -> row=lane, K [0..8)+[16..24);
// lanes 16-31 -> row=lane-16, K [8..16)+[24..32). Two b128 LDS loads.
static __device__ __forceinline__ v16bf ld_frag(const unsigned short* p) {
  Frag f;
  f.q[0] = *(const U4*)(p);
  f.q[1] = *(const U4*)(p + 16);
  return f.v;
}

// Core tiled GEMM: C(128x128,f32) += A(128xK bf16) * B(KxN f32->bf16).
// LDS double-buffered. A tiles stream global->LDS via async-load (ASYNCcnt);
// B tiles are register-staged (f32->bf16 k-pair packing) during the WMMA chain.
// A rows optionally gathered (gather[r] < 0 -> zero row, pre-zeroed in both
// buffers and never touched again). B staged transposed ([n][k]) so its
// fragments load identically to A's.
static __device__ __forceinline__ void gemm_core(
    const unsigned short* A, size_t lda, const int* gather,
    const float* B, int ldb, int n0, int K,
    unsigned short* lds, v8f acc[2][4])
{
  const int tid  = threadIdx.x;
  const int lane = tid & 31;
  const int wave = tid >> 5;
  const int wm   = wave & 3;     // 4 waves along M
  const int wn   = wave >> 2;    // 2 waves along N
  const int frow = lane & 15;
  const int koff = (lane >> 4) << 3;

  // ---- loop-invariant staging addresses ----
  const unsigned short* pA[2];   // global A source (nullptr -> zero row)
  int oA[2];                     // LDS store offset (elements)
#pragma unroll
  for (int it = 0; it < 2; ++it) {
    int idx = tid + it * 256;          // 0..511
    int r   = idx >> 2;                // 0..127
    int c   = (idx & 3) << 3;          // 0,8,16,24
    oA[it] = r * LDT + c;
    if (gather) {
      int g = gather[r];
      pA[it] = (g >= 0) ? (A + (size_t)g * lda + c) : nullptr;
    } else {
      pA[it] = A + (size_t)r * lda + c;
    }
  }
  const float* pB[2];            // even-k row of each k-pair unit
  int oBp[2];
#pragma unroll
  for (int it = 0; it < 2; ++it) {
    int idx = tid + it * 256;          // 0..511
    int kp  = idx >> 5;                // 0..15 (k pair)
    int nq  = (idx & 31) << 2;         // 0,4,...,124
    pB[it]  = B + (size_t)(2 * kp) * ldb + n0 + nq;
    oBp[it] = nq * LDT + 2 * kp;
  }
  const size_t stepB = (size_t)BK * ldb;

  // zero-fill gathered padding rows once, in both buffers (async never touches)
  if (gather) {
#pragma unroll
    for (int it = 0; it < 2; ++it)
      if (!pA[it]) {
        *(U4*)(lds + oA[it])         = U4{0u, 0u, 0u, 0u};
        *(U4*)(lds + BUFSZ + oA[it]) = U4{0u, 0u, 0u, 0u};
      }
  }

  // ---- prologue: stage tile 0 into buffer 0 ----
#pragma unroll
  for (int it = 0; it < 2; ++it)
    if (pA[it]) async_cp16(lds + oA[it], pA[it]);
  float4 b0[2], b1[2];
#pragma unroll
  for (int it = 0; it < 2; ++it) {
    b0[it] = *(const float4*)(pB[it]);
    b1[it] = *(const float4*)(pB[it] + ldb);
  }
#pragma unroll
  for (int it = 0; it < 2; ++it) {
    unsigned short* bd = lds + BBUF + oBp[it];
    *(unsigned*)(bd + 0 * LDT) = pack_bf16(b0[it].x, b1[it].x);
    *(unsigned*)(bd + 1 * LDT) = pack_bf16(b0[it].y, b1[it].y);
    *(unsigned*)(bd + 2 * LDT) = pack_bf16(b0[it].z, b1[it].z);
    *(unsigned*)(bd + 3 * LDT) = pack_bf16(b0[it].w, b1[it].w);
  }

  const int nk = K / BK;
  size_t kadv = 0;
  int buf = 0;
  for (int ki = 0; ki < nk; ++ki) {
    wait_async0();                         // our async A copies have landed
    __syncthreads();                       // everyone's staging visible
    const bool more = (ki + 1) < nk;
    unsigned short* d = lds + ((buf ^ 1) ? BUFSZ : 0);
    if (more) {                            // stage next tile into other buffer
      kadv += BK;
#pragma unroll
      for (int it = 0; it < 2; ++it)
        if (pA[it]) async_cp16(d + oA[it], pA[it] + kadv);
#pragma unroll
      for (int it = 0; it < 2; ++it) {
        const float* s = pB[it] + (size_t)(ki + 1) * stepB;
        b0[it] = *(const float4*)(s);
        b1[it] = *(const float4*)(s + ldb);
      }
      if (ki + 2 < nk)                     // warm L2 for the weight stream
        __builtin_prefetch(pB[0] + (size_t)(ki + 2) * stepB, 0, 1);
    }
    // ---- compute on lds[buf] ----
    const unsigned short* cb = lds + (buf ? BUFSZ : 0);
    v16bf af[2], bfr[4];
#pragma unroll
    for (int mi = 0; mi < 2; ++mi)
      af[mi] = ld_frag(cb + (wm * 32 + mi * 16 + frow) * LDT + koff);
#pragma unroll
    for (int ni = 0; ni < 4; ++ni)
      bfr[ni] = ld_frag(cb + BBUF + (wn * 64 + ni * 16 + frow) * LDT + koff);
#pragma unroll
    for (int mi = 0; mi < 2; ++mi)
#pragma unroll
      for (int ni = 0; ni < 4; ++ni)
        acc[mi][ni] = __builtin_amdgcn_wmma_f32_16x16x32_bf16(
            false, af[mi], false, bfr[ni], (short)0, acc[mi][ni], false, false);
    if (more) {                            // store packed B regs to other buffer
#pragma unroll
      for (int it = 0; it < 2; ++it) {
        unsigned short* bd = d + BBUF + oBp[it];
        *(unsigned*)(bd + 0 * LDT) = pack_bf16(b0[it].x, b1[it].x);
        *(unsigned*)(bd + 1 * LDT) = pack_bf16(b0[it].y, b1[it].y);
        *(unsigned*)(bd + 2 * LDT) = pack_bf16(b0[it].z, b1[it].z);
        *(unsigned*)(bd + 3 * LDT) = pack_bf16(b0[it].w, b1[it].w);
      }
    }
    buf ^= 1;
  }
}

// ---------------- kernel 1: x f32 -> bf16 (+ zero counters) ----------------
__global__ void __launch_bounds__(256) convert_kernel(const float* x,
                                                      unsigned short* xb,
                                                      int* counts) {
  int i = blockIdx.x * 256 + threadIdx.x;          // 4 elems each
  float4 v = ((const float4*)x)[i];
  uint2 p;
  p.x = pack_bf16(v.x, v.y);
  p.y = pack_bf16(v.z, v.w);
  ((uint2*)xb)[i] = p;
  if (blockIdx.x == 0 && threadIdx.x < NEXP) counts[threadIdx.x] = 0;
}

// ---------------- kernel 2: router (one wave per token) ----------------
__global__ void __launch_bounds__(256) router_kernel(
    const float* x, const float* gw, int* tk_idx, float* tk_w, int* counts,
    float* probs_part, float* logits_part) {
  const int tid = threadIdx.x, lane = tid & 31, wave = tid >> 5;
  const int t = blockIdx.x * 8 + wave;
  float acc[NEXP];
#pragma unroll
  for (int e = 0; e < NEXP; ++e) acc[e] = 0.f;

  for (int k = lane; k < DDIM; k += 32) {
    float xk = x[(size_t)t * DDIM + k];
    const float4* g = (const float4*)(gw + (size_t)k * NEXP);
    float4 g0 = g[0], g1 = g[1], g2 = g[2], g3 = g[3];
    acc[0] += xk * g0.x;  acc[1] += xk * g0.y;  acc[2]  += xk * g0.z;  acc[3]  += xk * g0.w;
    acc[4] += xk * g1.x;  acc[5] += xk * g1.y;  acc[6]  += xk * g1.z;  acc[7]  += xk * g1.w;
    acc[8] += xk * g2.x;  acc[9] += xk * g2.y;  acc[10] += xk * g2.z;  acc[11] += xk * g2.w;
    acc[12]+= xk * g3.x;  acc[13]+= xk * g3.y;  acc[14] += xk * g3.z;  acc[15] += xk * g3.w;
  }
#pragma unroll
  for (int o = 16; o >= 1; o >>= 1)
#pragma unroll
    for (int e = 0; e < NEXP; ++e) acc[e] += __shfl_xor(acc[e], o, 32);

  // softmax + top2 (redundant on all lanes, cheap)
  float mx = acc[0];
#pragma unroll
  for (int e = 1; e < NEXP; ++e) mx = fmaxf(mx, acc[e]);
  float pe[NEXP]; float sum = 0.f;
#pragma unroll
  for (int e = 0; e < NEXP; ++e) { pe[e] = __expf(acc[e] - mx); sum += pe[e]; }
  float inv = 1.f / sum;
#pragma unroll
  for (int e = 0; e < NEXP; ++e) pe[e] *= inv;
  int i0 = 0; float p0 = pe[0];
#pragma unroll
  for (int e = 1; e < NEXP; ++e) if (pe[e] > p0) { p0 = pe[e]; i0 = e; }
  int i1 = (i0 == 0) ? 1 : 0; float p1 = pe[i1];
#pragma unroll
  for (int e = 0; e < NEXP; ++e)
    if (e != i0 && pe[e] > p1) { p1 = pe[e]; i1 = e; }
  float wn = 1.f / (p0 + p1);

  __shared__ float sp[8][NEXP], sl[8][NEXP];
  if (lane == 0) {
    tk_idx[2 * t + 0] = i0;  tk_idx[2 * t + 1] = i1;
    tk_w [2 * t + 0] = p0 * wn;  tk_w [2 * t + 1] = p1 * wn;
    atomicAdd(&counts[i0], 1);
    atomicAdd(&counts[i1], 1);
#pragma unroll
    for (int e = 0; e < NEXP; ++e) { sp[wave][e] = pe[e]; sl[wave][e] = acc[e]; }
  }
  __syncthreads();
  if (tid < NEXP) {
    float a = 0.f, b = 0.f;
#pragma unroll
    for (int w = 0; w < 8; ++w) { a += sp[w][tid]; b += sl[w][tid]; }
    probs_part [blockIdx.x * NEXP + tid] = a;
    logits_part[blockIdx.x * NEXP + tid] = b;
  }
}

// ---------------- kernel 3: aux loss + tile scan (1 block) ----------------
__global__ void __launch_bounds__(256) finalize_kernel(
    const float* probs_part, const float* logits_part, const int* counts,
    int* base_slot, int* tile_expert, int* tile_slot0, int* num_tiles,
    int* cursor, int* slot_token, float* slot_weight, float* aux_out) {
  const int tid = threadIdx.x;
  __shared__ float sp[NEXP], sl[NEXP];
  if (tid < NEXP) {
    float a = 0.f, b = 0.f;
    for (int blk = 0; blk < ROUTER_BLOCKS; ++blk) {
      a += probs_part [blk * NEXP + tid];
      b += logits_part[blk * NEXP + tid];
    }
    sp[tid] = a; sl[tid] = b;
    cursor[tid] = 0;
  }
  __syncthreads();
  if (tid == 0) {
    float aux = 0.f;
    for (int e = 0; e < NEXP; ++e)
      aux += (sp[e] / (float)TTOT) * (sl[e] / (float)TTOT);
    aux_out[0] = aux * (float)NEXP;
    int t = 0, cum = 0;
    for (int e = 0; e < NEXP; ++e) {
      base_slot[e] = cum;
      int tiles = (counts[e] + BM - 1) / BM;
      for (int i = 0; i < tiles; ++i) { tile_expert[t] = e; tile_slot0[t] = cum + i * BM; ++t; }
      cum += tiles * BM;
    }
    num_tiles[0] = t;
  }
  for (int i = tid; i < MAXSLOTS; i += 256) { slot_token[i] = -1; slot_weight[i] = 0.f; }
}

// ---------------- kernel 4: token -> slot scatter ----------------
__global__ void __launch_bounds__(256) scatter_kernel(
    const int* tk_idx, const float* tk_w, const int* base_slot, int* cursor,
    int* slot_token, float* slot_weight, int* token_slot) {
  int t = blockIdx.x * 256 + threadIdx.x;
#pragma unroll
  for (int k = 0; k < 2; ++k) {
    int e = tk_idx[2 * t + k];
    int local = atomicAdd(&cursor[e], 1);
    int slot = base_slot[e] + local;
    slot_token[slot] = t;
    slot_weight[slot] = tk_w[2 * t + k];
    token_slot[2 * t + k] = slot;
  }
}

// ---------------- kernel 5: shared GEMM1 + SiLU ----------------
__global__ void __launch_bounds__(256) gemm1_shared_kernel(
    const unsigned short* xb, const float* w1, unsigned short* hout) {
  __shared__ unsigned short lds[2 * BUFSZ];
  const int s = blockIdx.z;
  const int n0 = blockIdx.x * BN;
  const int row0 = blockIdx.y * BM;
  v8f acc[2][4] = {};
  gemm_core(xb + (size_t)row0 * DDIM, DDIM, nullptr,
            w1 + (size_t)s * DDIM * HDIM, HDIM, n0, DDIM, lds, acc);
  const int tid = threadIdx.x, lane = tid & 31, wave = tid >> 5;
  const int wm = wave & 3, wnv = wave >> 2;
  const int colbase = s * HDIM + n0 + wnv * 64;
#pragma unroll
  for (int mi = 0; mi < 2; ++mi) {
    int m = row0 + wm * 32 + mi * 16 + ((lane >> 4) << 3);
#pragma unroll
    for (int ni = 0; ni < 4; ++ni) {
      int c = colbase + ni * 16 + (lane & 15);
#pragma unroll
      for (int v = 0; v < 8; ++v)
        hout[(size_t)(m + v) * (NSH * HDIM) + c] = f2bf(silu(acc[mi][ni][v]));
    }
  }
}

// ---------------- kernel 6: routed (gathered) GEMM1 + SiLU ----------------
__global__ void __launch_bounds__(256) gemm1_routed_kernel(
    const unsigned short* xb, const float* w1_all, unsigned short* hout,
    const int* slot_token, const int* tile_expert, const int* tile_slot0,
    const int* num_tiles) {
  const int tile = blockIdx.y;
  if (tile >= num_tiles[0]) return;                   // uniform per block
  __shared__ unsigned short lds[2 * BUFSZ];
  const int e = tile_expert[tile];
  const int slot0 = tile_slot0[tile];
  const int n0 = blockIdx.x * BN;
  v8f acc[2][4] = {};
  gemm_core(xb, DDIM, slot_token + slot0,
            w1_all + (size_t)e * DDIM * HDIM, HDIM, n0, DDIM, lds, acc);
  const int tid = threadIdx.x, lane = tid & 31, wave = tid >> 5;
  const int wm = wave & 3, wnv = wave >> 2;
  const int colbase = n0 + wnv * 64;
#pragma unroll
  for (int mi = 0; mi < 2; ++mi) {
    int m = slot0 + wm * 32 + mi * 16 + ((lane >> 4) << 3);
#pragma unroll
    for (int ni = 0; ni < 4; ++ni) {
      int c = colbase + ni * 16 + (lane & 15);
#pragma unroll
      for (int v = 0; v < 8; ++v)
        hout[(size_t)(m + v) * HDIM + c] = f2bf(silu(acc[mi][ni][v]));
    }
  }
}

// ---------------- kernel 7: shared GEMM2 (K = 2*H concat) -> d_out ----------
__global__ void __launch_bounds__(256) gemm2_shared_kernel(
    const unsigned short* h, const float* w2, float* out) {
  __shared__ unsigned short lds[2 * BUFSZ];
  const int n0 = blockIdx.x * BN;
  const int row0 = blockIdx.y * BM;
  v8f acc[2][4] = {};
  gemm_core(h + (size_t)row0 * (NSH * HDIM), NSH * HDIM, nullptr,
            w2, DDIM, n0, NSH * HDIM, lds, acc);
  const int tid = threadIdx.x, lane = tid & 31, wave = tid >> 5;
  const int wm = wave & 3, wnv = wave >> 2;
  const int colbase = n0 + wnv * 64;
#pragma unroll
  for (int mi = 0; mi < 2; ++mi) {
    int m = row0 + wm * 32 + mi * 16 + ((lane >> 4) << 3);
#pragma unroll
    for (int ni = 0; ni < 4; ++ni) {
      int c = colbase + ni * 16 + (lane & 15);
#pragma unroll
      for (int v = 0; v < 8; ++v)
        out[(size_t)(m + v) * DDIM + c] = acc[mi][ni][v];
    }
  }
}

// ---------------- kernel 8: routed GEMM2, weight-scaled, per-slot ----------
__global__ void __launch_bounds__(256) gemm2_routed_kernel(
    const unsigned short* hr, const float* w2_all, float* rout,
    const float* slot_weight, const int* tile_expert, const int* tile_slot0,
    const int* num_tiles) {
  const int tile = blockIdx.y;
  if (tile >= num_tiles[0]) return;
  __shared__ unsigned short lds[2 * BUFSZ];
  const int e = tile_expert[tile];
  const int slot0 = tile_slot0[tile];
  const int n0 = blockIdx.x * BN;
  v8f acc[2][4] = {};
  gemm_core(hr + (size_t)slot0 * HDIM, HDIM, nullptr,
            w2_all + (size_t)e * HDIM * DDIM, DDIM, n0, HDIM, lds, acc);
  const int tid = threadIdx.x, lane = tid & 31, wave = tid >> 5;
  const int wm = wave & 3, wnv = wave >> 2;
  const int colbase = n0 + wnv * 64;
#pragma unroll
  for (int mi = 0; mi < 2; ++mi) {
    int m = slot0 + wm * 32 + mi * 16 + ((lane >> 4) << 3);
#pragma unroll
    for (int ni = 0; ni < 4; ++ni) {
      int c = colbase + ni * 16 + (lane & 15);
#pragma unroll
      for (int v = 0; v < 8; ++v)
        rout[(size_t)(m + v) * DDIM + c] = acc[mi][ni][v] * slot_weight[m + v];
    }
  }
}

// ---------------- kernel 9: combine routed slots into output ----------------
__global__ void __launch_bounds__(256) combine_kernel(
    float* out, const float* rout, const int* token_slot) {
  const int t = blockIdx.x;
  const int s0 = token_slot[2 * t + 0];
  const int s1 = token_slot[2 * t + 1];
  const float4* r0 = (const float4*)(rout + (size_t)s0 * DDIM);
  const float4* r1 = (const float4*)(rout + (size_t)s1 * DDIM);
  float4* o = (float4*)(out + (size_t)t * DDIM);
  for (int i = threadIdx.x; i < DDIM / 4; i += 256) {
    float4 a = o[i], u = r0[i], v = r1[i];
    a.x += u.x + v.x; a.y += u.y + v.y; a.z += u.z + v.z; a.w += u.w + v.w;
    o[i] = a;
  }
}

// ---------------- host side ----------------
extern "C" void kernel_launch(void* const* d_in, const int* in_sizes, int n_in,
                              void* d_out, int out_size, void* d_ws, size_t ws_size,
                              hipStream_t stream) {
  const float* x         = (const float*)d_in[0];
  const float* gate_w    = (const float*)d_in[1];
  const float* shared_w1 = (const float*)d_in[2];
  const float* shared_w2 = (const float*)d_in[3];   // [2][1408][2048] == [2816][2048]
  const float* expert_w1 = (const float*)d_in[4];
  const float* expert_w2 = (const float*)d_in[5];
  float* out = (float*)d_out;

  char* p = (char*)d_ws;
  auto alloc = [&](size_t bytes) -> char* {
    char* r = p; p += (bytes + 255) & ~(size_t)255; return r;
  };
  unsigned short* xb   = (unsigned short*)alloc((size_t)TTOT * DDIM * 2);       // 16.8 MB
  unsigned short* h_sh = (unsigned short*)alloc((size_t)TTOT * NSH * HDIM * 2); // 23.1 MB
  unsigned short* h_rt = (unsigned short*)alloc((size_t)MAXSLOTS * HDIM * 2);   // 28.8 MB
  float* rout          = (float*)alloc((size_t)MAXSLOTS * DDIM * 4);            // 83.9 MB
  int*   tk_idx      = (int*)  alloc((size_t)TTOT * 2 * 4);
  float* tk_w        = (float*)alloc((size_t)TTOT * 2 * 4);
  int*   token_slot  = (int*)  alloc((size_t)TTOT * 2 * 4);
  int*   slot_token  = (int*)  alloc((size_t)MAXSLOTS * 4);
  float* slot_weight = (float*)alloc((size_t)MAXSLOTS * 4);
  int*   counts      = (int*)  alloc(64);
  int*   cursor      = (int*)  alloc(64);
  int*   base_slot   = (int*)  alloc(64);
  int*   tile_expert = (int*)  alloc(128 * 4);
  int*   tile_slot0  = (int*)  alloc(128 * 4);
  int*   num_tiles   = (int*)  alloc(64);
  float* probs_part  = (float*)alloc((size_t)ROUTER_BLOCKS * NEXP * 4);
  float* logits_part = (float*)alloc((size_t)ROUTER_BLOCKS * NEXP * 4);

  convert_kernel<<<(TTOT * DDIM) / 1024, 256, 0, stream>>>(x, xb, counts);
  router_kernel<<<ROUTER_BLOCKS, 256, 0, stream>>>(x, gate_w, tk_idx, tk_w,
                                                   counts, probs_part, logits_part);
  finalize_kernel<<<1, 256, 0, stream>>>(probs_part, logits_part, counts,
                                         base_slot, tile_expert, tile_slot0,
                                         num_tiles, cursor, slot_token, slot_weight,
                                         out + (size_t)TTOT * DDIM);
  scatter_kernel<<<TTOT / 256, 256, 0, stream>>>(tk_idx, tk_w, base_slot, cursor,
                                                 slot_token, slot_weight, token_slot);
  gemm1_shared_kernel<<<dim3(HDIM / BN, TTOT / BM, NSH), 256, 0, stream>>>(
      xb, shared_w1, h_sh);
  gemm1_routed_kernel<<<dim3(HDIM / BN, MAXTILES), 256, 0, stream>>>(
      xb, expert_w1, h_rt, slot_token, tile_expert, tile_slot0, num_tiles);
  gemm2_shared_kernel<<<dim3(DDIM / BN, TTOT / BM), 256, 0, stream>>>(
      h_sh, shared_w2, out);
  gemm2_routed_kernel<<<dim3(DDIM / BN, MAXTILES), 256, 0, stream>>>(
      h_rt, expert_w2, rout, slot_weight, tile_expert, tile_slot0, num_tiles);
  combine_kernel<<<TTOT, 256, 0, stream>>>(out, rout, token_slot);
}
